// Generator_80427557584920
// MI455X (gfx1250) — matrix operand order repaired and checked by reading, not compiled, not run on hardware
//
#include <hip/hip_runtime.h>

// ---------------------------------------------------------------------------
// StyleGAN3 block: modulated 3x3 conv (implicit GEMM, bf16 WMMA) + fused
// filtered_lrelu (up2 FIR -> lrelu -> clamp -> down2 FIR) for MI455X/gfx1250.
// ---------------------------------------------------------------------------

#define B_      4
#define CIN_    256
#define COUT_   256
#define H_      128
#define W_      128
#define SD_     512
#define HO_     130          // conv output spatial (H + K - 1)
#define P_TOT   (HO_*HO_)    // 16900 flattened spatial
#define NTAPS   12
#define PADH_   134          // padded bf16 input rows  (covers py+ky <= 133)
#define PADW_   136          // padded bf16 input row stride
#define PLANE_  (PADH_*PADW_)

typedef __attribute__((ext_vector_type(16))) __bf16 v16bf;
typedef __attribute__((ext_vector_type(8)))  float  v8f;

// ---------------------------------------------------------------- kernel 1 --
// s[b][cin] = style[b] . mod_weight[cin] / sqrt(512) + mod_bias[cin]
__global__ __launch_bounds__(256) void k_modulate(
    const float* __restrict__ style, const float* __restrict__ modw,
    const float* __restrict__ modb, float* __restrict__ s)
{
  __shared__ float st[SD_];
  const int b = blockIdx.x, cin = threadIdx.x;
  for (int d = threadIdx.x; d < SD_; d += 256) st[d] = style[b*SD_ + d];
  __syncthreads();
  const float* wr = modw + (size_t)cin * SD_;
  float acc = 0.f;
  #pragma unroll 8
  for (int d = 0; d < SD_; ++d) acc += st[d] * wr[d];
  s[b*CIN_ + cin] = acc * 0.04419417382415922f + modb[cin];
}

// ---------------------------------------------------------------- kernel 2 --
// Per (b,cout): demod = rsqrt(sum((w*s)^2)+eps); emit bf16 weights laid out
// as [b][tap][cout][cin]  (cin contiguous -> coalesced, 16B GEMM A loads).
__global__ __launch_bounds__(256) void k_demod(
    const float* __restrict__ weight, const float* __restrict__ s,
    __bf16* __restrict__ wmod)
{
  __shared__ float red[256];
  const int bc = blockIdx.x;
  const int b = bc >> 8, cout = bc & 255;
  const int cin = threadIdx.x;

  const float sv = s[b*CIN_ + cin];
  const float* wp = weight + ((size_t)cout*CIN_ + cin) * 9;
  float v[9], ss = 0.f;
  #pragma unroll
  for (int t = 0; t < 9; ++t) { v[t] = wp[t] * sv; ss += v[t]*v[t]; }

  red[cin] = ss; __syncthreads();
  for (int off = 128; off > 0; off >>= 1) {
    if (cin < off) red[cin] += red[cin + off];
    __syncthreads();
  }
  const float demod = rsqrtf(red[0] + 1e-8f);

  #pragma unroll
  for (int t = 0; t < 9; ++t)
    wmod[(((size_t)(b*9 + t)*COUT_ + cout) << 8) + cin] = (__bf16)(v[t] * demod);
}

// ---------------------------------------------------------------- kernel 2b -
// Zero-padded bf16 copy of input: [b*cin][134][136]; interior at (+2,+2).
// Removes all predicates / converts from the GEMM hot loop.
__global__ __launch_bounds__(256) void k_pad(
    const float* __restrict__ input, __bf16* __restrict__ inpad)
{
  const int plane = blockIdx.x;                 // b*CIN + cin
  const float* src = input + (size_t)plane * (H_*W_);
  __bf16* dst = inpad + (size_t)plane * PLANE_;
  for (int i = threadIdx.x; i < PLANE_; i += 256) {
    const int yy = i / PADW_, xx = i % PADW_;
    float v = 0.f;
    if ((unsigned)(yy - 2) < (unsigned)H_ && (unsigned)(xx - 2) < (unsigned)W_)
      v = src[(yy - 2)*W_ + (xx - 2)];
    dst[i] = (__bf16)v;
  }
}

// ---------------------------------------------------------------- kernel 3 --
// Implicit-GEMM conv: per batch, M=Cout(256), N=130*130, K=Cin*9 (2304).
// Block tile 64(M) x 256(N); 8 waves as 2(M)x4(N), wave tile 32x64:
// 2 A-frags + 4 B-frags -> 8 WMMAs per K-iter (each B-frag feeds 2 WMMAs).
// Ping-pong LDS (one barrier / K-iter); fragments loaded as raw uint4 pairs.
__global__ __launch_bounds__(256) void k_conv_wmma(
    const __bf16* __restrict__ inpad, const __bf16* __restrict__ wmod,
    const float* __restrict__ bias, float* __restrict__ convout)
{
  __shared__ __bf16 As[2][64][40];    // [m][k], 80B row stride (bank-friendly)
  __shared__ __bf16 Bs[2][256][40];   // [p][k]

  const int tid  = threadIdx.x;
  const int lane = tid & 31;
  const int wave = tid >> 5;
  const int wm   = wave & 1;          // 32-row cout sub-tile
  const int wn   = wave >> 1;         // 64-col spatial sub-tile

  const int b        = blockIdx.z;
  const int coutBase = blockIdx.y * 64;
  const int pBase    = blockIdx.x * 256;

  const v8f zero = {0.f,0.f,0.f,0.f,0.f,0.f,0.f,0.f};
  v8f acc[2][4];
  #pragma unroll
  for (int i = 0; i < 2; ++i)
    #pragma unroll
    for (int nt = 0; nt < 4; ++nt) acc[i][nt] = zero;

  // A-fill: 4 threads cover one m-row's 32 k's (64B contiguous global)
  const int mA  = tid >> 2;
  const int kcA = (tid & 3) * 8;
  // B-fill: thread owns spatial p = tid, covers all 32 k's
  const int pg  = pBase + tid;        // may overflow P_TOT; pad covers it
  const int py  = pg / HO_;
  const int px  = pg % HO_;

  // fragment addressing (ISA VGPR layouts)
  const int kA    = (lane & 16) ? 8 : 0;   // A: lanes<16 K{0..7,16..23}
  const int kB    = (lane & 16) ? 8 : 0;   // B: lanes<16 K{0..7,16..23}
  const int nbase = wn*64 + (lane & 15);

  auto stage = [&](int kk, int buf) {
    const int tap  = kk >> 3;
    const int cin0 = (kk & 7) << 5;
    // A tile: wmod[b][tap][coutBase+mA][cin0+kcA .. +7]
    const __bf16* gp = wmod + (((size_t)(b*9 + tap)*COUT_ + coutBase + mA) << 8)
                            + cin0 + kcA;
    *(uint4*)&As[buf][mA][kcA] = *(const uint4*)gp;
    // B tile: padded bf16 input, no predicates, no converts
    const int rowoff = (py + tap/3)*PADW_ + px + (tap%3);
    const __bf16* ip = inpad + (size_t)(b*CIN_ + cin0)*PLANE_ + rowoff;
    #pragma unroll
    for (int k = 0; k < 32; ++k)
      Bs[buf][tid][k] = ip[(size_t)k*PLANE_];
  };

  auto compute = [&](int buf) {
    union Frag { uint4 q[2]; v16bf v; };
    Frag ua[2], ub;
    #pragma unroll
    for (int i = 0; i < 2; ++i) {
      const int mr = wm*32 + i*16 + (lane & 15);
      ua[i].q[0] = *(const uint4*)&As[buf][mr][kA];
      ua[i].q[1] = *(const uint4*)&As[buf][mr][kA + 16];
    }
    #pragma unroll
    for (int nt = 0; nt < 4; ++nt) {
      ub.q[0] = *(const uint4*)&Bs[buf][nbase + nt*16][kB];
      ub.q[1] = *(const uint4*)&Bs[buf][nbase + nt*16][kB + 16];
      #pragma unroll
      for (int i = 0; i < 2; ++i)
        acc[i][nt] = __builtin_amdgcn_wmma_f32_16x16x32_bf16(
            false, ua[i].v, false, ub.v, (short)0, acc[i][nt], false, false);
    }
  };

  stage(0, 0);
  for (int kk = 0; kk < 72; ++kk) {
    __syncthreads();
    if (kk + 1 < 72) stage(kk + 1, (kk + 1) & 1);
    compute(kk & 1);
  }

  // epilogue: C layout VGPR r -> M = r (+8 for lanes>=16), N = lane&15
  const int mSub = (lane & 16) ? 8 : 0;
  const int nSub = lane & 15;
  #pragma unroll
  for (int nt = 0; nt < 4; ++nt) {
    const int pout = pBase + wn*64 + nt*16 + nSub;
    if (pout >= P_TOT) continue;
    #pragma unroll
    for (int i = 0; i < 2; ++i) {
      #pragma unroll
      for (int r = 0; r < 8; ++r) {
        const int m = coutBase + wm*32 + i*16 + r + mSub;
        convout[((size_t)(b*COUT_ + m))*P_TOT + pout] = acc[i][nt][r] + bias[m];
      }
    }
  }
}

// ---------------------------------------------------------------- kernel 4 --
// Fused filtered_lrelu per (b, c, 26x26 output tile). All intermediates
// (36x36 halo, 62-wide up grids) live in LDS; nothing up-sampled hits HBM.
__global__ __launch_bounds__(256) void k_flrelu(
    const float* __restrict__ conv, const float* __restrict__ upf,
    const float* __restrict__ dnf, float* __restrict__ out)
{
  __shared__ float fu[NTAPS], fd[NTAPS];
  __shared__ float convT[36][36];
  __shared__ float upv[62][36];
  __shared__ float uph[62][62];
  __shared__ float dnv[26][62];

  const int tid = threadIdx.x;
  const int ty = blockIdx.x / 5, tx = blockIdx.x % 5;
  const int y0 = ty * 26, x0 = tx * 26;
  const int c = blockIdx.y, b = blockIdx.z;
  const float* cbase = conv + ((size_t)(b*COUT_ + c)) * P_TOT;

  if (tid < NTAPS) { fu[tid] = upf[tid] * 2.0f; fd[tid] = dnf[tid]; }

  for (int i = tid; i < 36*36; i += 256) {
    const int rl = i / 36, cl = i % 36;
    const int gy = y0 - 5 + rl, gx = x0 - 5 + cl;
    float v = 0.f;
    if ((unsigned)gy < (unsigned)HO_ && (unsigned)gx < (unsigned)HO_)
      v = cbase[gy*HO_ + gx];
    convT[rl][cl] = v;
  }
  __syncthreads();

  // vertical x2 up-FIR (zero-stuffed: only taps matching parity contribute)
  for (int i = tid; i < 62*36; i += 256) {
    const int lu = i / 36, cl = i % 36;
    const int yu = 2*y0 - 5 + lu;
    float acc = 0.f;
    if ((unsigned)yu < 260u) {
      for (int t = (yu & 1); t < NTAPS; t += 2) {
        const int r = (yu - 6 + t) >> 1;
        if ((unsigned)r < (unsigned)HO_) acc += fu[t] * convT[r - (y0 - 5)][cl];
      }
    }
    upv[lu][cl] = acc;
  }
  __syncthreads();

  // horizontal x2 up-FIR + lrelu*sqrt(2) + clamp
  for (int i = tid; i < 62*62; i += 256) {
    const int lu = i / 62, lx = i % 62;
    const int xu = 2*x0 - 5 + lx;
    float acc = 0.f;
    if ((unsigned)xu < 260u) {
      for (int t = (xu & 1); t < NTAPS; t += 2) {
        const int cX = (xu - 6 + t) >> 1;
        if ((unsigned)cX < (unsigned)HO_) acc += fu[t] * upv[lu][cX - (x0 - 5)];
      }
    }
    acc = (acc >= 0.f ? acc : 0.2f * acc) * 1.4142135623730951f;
    acc = fminf(fmaxf(acc, -256.f), 256.f);
    uph[lu][lx] = acc;
  }
  __syncthreads();

  // vertical /2 down-FIR (pad 5): local up row = 2*ld + t
  for (int i = tid; i < 26*62; i += 256) {
    const int ld = i / 62, lx = i % 62;
    float acc = 0.f;
    #pragma unroll
    for (int t = 0; t < NTAPS; ++t) acc += fd[t] * uph[2*ld + t][lx];
    dnv[ld][lx] = acc;
  }
  __syncthreads();

  // horizontal /2 down-FIR -> output
  for (int i = tid; i < 26*26; i += 256) {
    const int ld = i / 26, lo = i % 26;
    float acc = 0.f;
    #pragma unroll
    for (int t = 0; t < NTAPS; ++t) acc += fd[t] * dnv[ld][2*lo + t];
    out[((size_t)(b*COUT_ + c))*P_TOT + (y0 + ld)*HO_ + (x0 + lo)] = acc;
  }
}

// ---------------------------------------------------------------- launcher --
extern "C" void kernel_launch(void* const* d_in, const int* in_sizes, int n_in,
                              void* d_out, int out_size, void* d_ws, size_t ws_size,
                              hipStream_t stream) {
  const float* input  = (const float*)d_in[0];
  const float* style  = (const float*)d_in[1];
  const float* weight = (const float*)d_in[2];
  const float* modw   = (const float*)d_in[3];
  const float* modb   = (const float*)d_in[4];
  const float* bias   = (const float*)d_in[5];
  const float* upf    = (const float*)d_in[6];
  const float* dnf    = (const float*)d_in[7];
  float* out = (float*)d_out;

  // workspace layout:
  //   s (4 KB) | wmod bf16 (4.5 MB) | inpad bf16 (37.3 MB) | conv f32 (69 MB)
  char* ws = (char*)d_ws;
  size_t off = 0;
  float*  s       = (float*)(ws + off);  off += 4096;
  __bf16* wmod    = (__bf16*)(ws + off); off += (size_t)B_*9*COUT_*CIN_*2;
  __bf16* inpad   = (__bf16*)(ws + off); off += (size_t)B_*CIN_*PLANE_*2;
  float*  convout = (float*)(ws + off);

  k_modulate<<<dim3(B_), dim3(256), 0, stream>>>(style, modw, modb, s);
  k_demod<<<dim3(B_*COUT_), dim3(256), 0, stream>>>(weight, s, wmod);
  k_pad<<<dim3(B_*CIN_), dim3(256), 0, stream>>>(input, inpad);
  k_conv_wmma<<<dim3((P_TOT + 255)/256, COUT_/64, B_), dim3(256), 0, stream>>>(
      inpad, wmod, bias, convout);
  k_flrelu<<<dim3(25, COUT_, B_), dim3(256), 0, stream>>>(convout, upf, dnf, out);
}
